// SelfAttentionSAG_16363825397934
// MI455X (gfx1250) — matrix operand order repaired and checked by reading, not compiled, not run on hardware
//
#include <hip/hip_runtime.h>
#include <math.h>

#define NN 8192
#define DINC 128
#define DH 64
#define DOUTC 32
#define KKEEP 4096

typedef __attribute__((ext_vector_type(16))) __bf16 v16bf;
typedef __attribute__((ext_vector_type(8)))  float  v8f;

struct __align__(16) U128 { unsigned x, y, z, w; };
typedef U128 __attribute__((may_alias)) U128a;
struct __align__(32) U256 { U128 lo, hi; };

static __device__ __forceinline__ v16bf pack_frag(U128 lo, U128 hi) {
  U256 t; t.lo = lo; t.hi = hi;
  return __builtin_bit_cast(v16bf, t);
}
static __device__ __forceinline__ v8f zero8() {
  v8f z;
#pragma unroll
  for (int i = 0; i < 8; ++i) z[i] = 0.0f;
  return z;
}
// A-fragment (16x32 bf16, MxK) from row-major source. lane m = row, per-lane
// elements are K = k0+8*half+0..7 and K = k0+16+8*half+0..7 (two 16B chunks).
static __device__ __forceinline__ v16bf afrag_bf16(const __bf16* base, int stride,
                                                   int m, int half, int k0) {
  const __bf16* p = base + m * stride + k0 + 8 * half;
  U128 lo = *reinterpret_cast<const U128a*>(p);
  U128 hi = *reinterpret_cast<const U128a*>(p + 16);
  return pack_frag(lo, hi);
}
// B-fragment (32x16 bf16, KxN) from N-major (transposed) source Bt[n][k].
// lane n = column, per-lane elements are K = k0+8*half+0..7 and K = k0+16+8*half+0..7 (two 16B chunks).
static __device__ __forceinline__ v16bf bfrag_bf16(const __bf16* bt, int stride,
                                                   int n, int half, int k0) {
  const __bf16* p = bt + n * stride + k0 + 8 * half;
  U128 lo = *reinterpret_cast<const U128a*>(p);
  U128 hi = *reinterpret_cast<const U128a*>(p + 16);
  return pack_frag(lo, hi);
}
// A-fragment converting f32 source to bf16 on the fly.
static __device__ __forceinline__ v16bf afrag_f32(const float* base, int stride,
                                                  int m, int half, int k0) {
  const float* p = base + m * stride + k0 + 8 * half;
  v16bf a;
#pragma unroll
  for (int i = 0; i < 8; ++i) a[i] = (__bf16)p[i];
#pragma unroll
  for (int i = 0; i < 8; ++i) a[8 + i] = (__bf16)p[16 + i];
  return a;
}
#define WMMA_BF16(A, B, C) \
  __builtin_amdgcn_wmma_f32_16x16x32_bf16(false, (A), false, (B), (short)0, (C), false, false)

// ---------------- Kernel 1: Q/K/V projections (bf16 out, 1/8 folded into Q) --
__global__ __launch_bounds__(256) void qkv_kernel(
    const float* __restrict__ x,
    const float* __restrict__ Wq, const float* __restrict__ bq,
    const float* __restrict__ Wk, const float* __restrict__ bk,
    const float* __restrict__ Wv, const float* __restrict__ bv,
    __bf16* __restrict__ Qb, __bf16* __restrict__ Kb, __bf16* __restrict__ VbT) {
  __shared__ __align__(16) __bf16 wt[3][DH * DINC];  // Wt[n][k] (transposed, bf16)
  const int tid = threadIdx.x;
#pragma unroll
  for (int t = 0; t < 3; ++t) {
    const float* Wsrc = (t == 0) ? Wq : ((t == 1) ? Wk : Wv);
    for (int e = tid; e < DINC * DH; e += 256) {
      int k = e >> 6, n = e & 63;
      wt[t][n * DINC + k] = (__bf16)Wsrc[e];
    }
  }
  __syncthreads();
  const int wave = tid >> 5, lane = tid & 31;
  const int m = lane & 15, half = lane >> 4;
  const int row0 = blockIdx.x * 64;
  for (int jj = 0; jj < 6; ++jj) {
    int job = wave * 6 + jj;
    int t = job >> 4;
    int tile = job & 15;
    int mt = tile >> 2, nt = tile & 3;
    v8f acc = zero8();
#pragma unroll
    for (int kk = 0; kk < 4; ++kk) {
      v16bf a = afrag_f32(x + (row0 + mt * 16) * DINC, DINC, m, half, kk * 32);
      v16bf b = bfrag_bf16(&wt[t][nt * 16 * DINC], DINC, m, half, kk * 32);
      acc = WMMA_BF16(a, b, acc);
    }
    const float* bp = (t == 0) ? bq : ((t == 1) ? bk : bv);
    float bias = bp[nt * 16 + m];
    float scal = (t == 0) ? 0.125f : 1.0f;  // fold 1/sqrt(DH) into Q
#pragma unroll
    for (int r = 0; r < 8; ++r) {
      int grow = row0 + mt * 16 + r + 8 * half;
      int col = nt * 16 + m;
      float v = (acc[r] + bias) * scal;
      if (t == 0)      Qb[grow * DH + col] = (__bf16)v;
      else if (t == 1) Kb[grow * DH + col] = (__bf16)v;
      else             VbT[col * NN + grow] = (__bf16)v;  // V transposed [d][row]
    }
  }
}

// ---------------- Kernel 2: fused flash attention + exact edge stats ---------
// 4 waves x 16 query rows; streams 128-key K/V tiles via async LDS loads.
__global__ __launch_bounds__(128) void attn_kernel(
    const __bf16* __restrict__ Qb, const __bf16* __restrict__ Kb,
    const __bf16* __restrict__ VbT, float* __restrict__ h,
    float* __restrict__ dinv, int* __restrict__ jstar, int* __restrict__ eflag) {
  __shared__ __align__(16) __bf16 kl[128 * DH];      // K tile   [key][d]
  __shared__ __align__(16) __bf16 vt[DH * 128];      // V tile   [d][key]
  __shared__ __align__(16) __bf16 pl[4][16 * 128];   // per-wave P scratch
  const int tid = threadIdx.x;
  const int wave = tid >> 5, lane = tid & 31;
  const int m = lane & 15, half = lane >> 4;
  const int qrow0 = blockIdx.x * 64 + wave * 16;

  v16bf qf0 = afrag_bf16(Qb + qrow0 * DH, DH, m, half, 0);
  v16bf qf1 = afrag_bf16(Qb + qrow0 * DH, DH, m, half, 32);

  v8f O[4];
  float mrow[8], Zrow[8];
  int aidx[8];
#pragma unroll
  for (int nt = 0; nt < 4; ++nt) O[nt] = zero8();
#pragma unroll
  for (int r = 0; r < 8; ++r) { mrow[r] = -INFINITY; Zrow[r] = 0.0f; aidx[r] = 0; }

  unsigned kbase = (unsigned)(uintptr_t)&kl[0];
  unsigned vbase = (unsigned)(uintptr_t)&vt[0];

  for (int blk = 0; blk < NN / 128; ++blk) {
    // --- async global -> LDS tile loads (tracked by ASYNCcnt) ---
#pragma unroll
    for (int it = 0; it < 8; ++it) {
      int chunk = tid + it * 128;               // 1024 16B chunks of K tile
      int row = chunk >> 3, c8 = chunk & 7;
      unsigned goff = (unsigned)(((blk * 128 + row) * DH + c8 * 8) * 2);
      unsigned laddr = kbase + (unsigned)((row * DH + c8 * 8) * 2);
      asm volatile("global_load_async_to_lds_b128 %0, %1, %2"
                   :: "v"(laddr), "v"(goff), "s"(Kb) : "memory");
    }
#pragma unroll
    for (int it = 0; it < 8; ++it) {
      int chunk = tid + it * 128;               // 1024 16B chunks of Vt tile
      int d = chunk >> 4, c = chunk & 15;
      unsigned goff = (unsigned)((d * NN + blk * 128 + c * 8) * 2);
      unsigned laddr = vbase + (unsigned)((d * 128 + c * 8) * 2);
      asm volatile("global_load_async_to_lds_b128 %0, %1, %2"
                   :: "v"(laddr), "v"(goff), "s"(VbT) : "memory");
    }
    asm volatile("s_wait_asynccnt 0" ::: "memory");
    __syncthreads();

    // --- S = Q * K^T for 8 sub-tiles of 16 keys ---
    v8f S[8];
#pragma unroll
    for (int kb = 0; kb < 8; ++kb) {
      v16bf b0 = bfrag_bf16(kl + kb * 16 * DH, DH, m, half, 0);
      v16bf b1 = bfrag_bf16(kl + kb * 16 * DH, DH, m, half, 32);
      v8f s = zero8();
      s = WMMA_BF16(qf0, b0, s);
      s = WMMA_BF16(qf1, b1, s);
      S[kb] = s;
    }
    // --- block row max + argmax (ties -> smallest index) ---
    float bm[8]; int bi[8];
#pragma unroll
    for (int r = 0; r < 8; ++r) { bm[r] = S[0][r]; bi[r] = blk * 128 + m; }
#pragma unroll
    for (int kb = 1; kb < 8; ++kb) {
#pragma unroll
      for (int r = 0; r < 8; ++r) {
        float v = S[kb][r]; int idx = blk * 128 + kb * 16 + m;
        if (v > bm[r] || (v == bm[r] && idx < bi[r])) { bm[r] = v; bi[r] = idx; }
      }
    }
#pragma unroll
    for (int r = 0; r < 8; ++r) {
      float v = bm[r]; int i = bi[r];
#pragma unroll
      for (int d = 1; d < 16; d <<= 1) {
        float ov = __shfl_xor(v, d, 32);
        int oi = __shfl_xor(i, d, 32);
        if (ov > v || (ov == v && oi < i)) { v = ov; i = oi; }
      }
      bm[r] = v; bi[r] = i;
    }
    // --- online softmax state update + rescale O/Z ---
#pragma unroll
    for (int r = 0; r < 8; ++r) {
      float mold = mrow[r];
      float mn = fmaxf(mold, bm[r]);
      if (bm[r] > mold) aidx[r] = bi[r];    // first-occurrence argmax
      float sc = __expf(mold - mn);
      mrow[r] = mn;
      Zrow[r] *= sc;
#pragma unroll
      for (int nt = 0; nt < 4; ++nt) O[nt][r] *= sc;
    }
    // --- P = exp(S - m); Z += rowsum(P); stash P (bf16) for layout transpose ---
    float zp[8];
#pragma unroll
    for (int r = 0; r < 8; ++r) zp[r] = 0.0f;
#pragma unroll
    for (int kb = 0; kb < 8; ++kb) {
#pragma unroll
      for (int r = 0; r < 8; ++r) {
        float p = __expf(S[kb][r] - mrow[r]);
        zp[r] += p;
        pl[wave][(r + 8 * half) * 128 + kb * 16 + m] = (__bf16)p;
      }
    }
#pragma unroll
    for (int r = 0; r < 8; ++r) {
      float z = zp[r];
#pragma unroll
      for (int d = 1; d < 16; d <<= 1) z += __shfl_xor(z, d, 32);
      Zrow[r] += z;
    }
    // --- O += P @ V ---
#pragma unroll
    for (int kp = 0; kp < 4; ++kp) {
      v16bf pf = afrag_bf16(&pl[wave][0], 128, m, half, kp * 32);
#pragma unroll
      for (int nt = 0; nt < 4; ++nt) {
        v16bf vf = bfrag_bf16(vt + nt * 16 * 128, 128, m, half, kp * 32);
        O[nt] = WMMA_BF16(pf, vf, O[nt]);
      }
    }
    __syncthreads();  // tiles consumed before next block's async loads land
  }
  // --- epilogue: h = O / Z, per-row edge stats (max attn = 1/Z) ---
#pragma unroll
  for (int nt = 0; nt < 4; ++nt)
#pragma unroll
    for (int r = 0; r < 8; ++r)
      h[(qrow0 + r + 8 * half) * DH + nt * 16 + m] = O[nt][r] * (1.0f / Zrow[r]);
  if (m == 0) {
#pragma unroll
    for (int r = 0; r < 8; ++r) {
      int row = qrow0 + r + 8 * half;
      int e = (Zrow[r] < 2.0f) && (aidx[r] != row);
      dinv[row] = rsqrtf(1.0f + (float)e);
      jstar[row] = aidx[r];
      eflag[row] = e;
    }
  }
}

// ---------------- Kernel 3a: u = h @ gcn_W ----------------------------------
__global__ __launch_bounds__(256) void gcn_mm_kernel(
    const float* __restrict__ h, const float* __restrict__ gcn_W,
    float* __restrict__ u) {
  __shared__ __align__(16) __bf16 wt[DH * DH];  // Wt[n][k]
  const int tid = threadIdx.x;
  for (int e = tid; e < DH * DH; e += 256) {
    int k = e >> 6, n = e & 63;
    wt[n * DH + k] = (__bf16)gcn_W[e];
  }
  __syncthreads();
  const int wave = tid >> 5, lane = tid & 31;
  const int m = lane & 15, half = lane >> 4;
  const int row0 = blockIdx.x * 128 + wave * 16;
  v16bf a0 = afrag_f32(h + row0 * DH, DH, m, half, 0);
  v16bf a1 = afrag_f32(h + row0 * DH, DH, m, half, 32);
#pragma unroll
  for (int nt = 0; nt < 4; ++nt) {
    v8f acc = zero8();
    acc = WMMA_BF16(a0, bfrag_bf16(wt + nt * 16 * DH, DH, m, half, 0), acc);
    acc = WMMA_BF16(a1, bfrag_bf16(wt + nt * 16 * DH, DH, m, half, 32), acc);
#pragma unroll
    for (int r = 0; r < 8; ++r)
      u[(row0 + r + 8 * half) * DH + nt * 16 + m] = acc[r];
  }
}

// ---------------- Kernel 3b: g = An@u + b (exact sparse) and s0 = g . sW -----
__global__ __launch_bounds__(256) void gprop_kernel(
    const float* __restrict__ u, const float* __restrict__ dinv,
    const int* __restrict__ jstar, const int* __restrict__ eflag,
    const float* __restrict__ gcn_b, const float* __restrict__ score_W,
    float* __restrict__ g, float* __restrict__ s0) {
  const int wave = threadIdx.x >> 5, lane = threadIdx.x & 31;
  const int row = blockIdx.x * 8 + wave;
  float di = dinv[row];
  int j = jstar[row];
  float cross = eflag[row] ? di * dinv[j] : 0.0f;
  float selfc = di * di;
  float g0 = selfc * u[row * DH + lane]      + cross * u[j * DH + lane]      + gcn_b[lane];
  float g1 = selfc * u[row * DH + lane + 32] + cross * u[j * DH + lane + 32] + gcn_b[lane + 32];
  g[row * DH + lane] = g0;
  g[row * DH + lane + 32] = g1;
  float part = g0 * score_W[lane] + g1 * score_W[lane + 32];
#pragma unroll
  for (int d = 1; d < 32; d <<= 1) part += __shfl_xor(part, d, 32);
  if (lane == 0) s0[row] = part;
}

// ---------------- Kernel 3c: score = tanh(An@s0 + b) -------------------------
__global__ __launch_bounds__(256) void score_kernel(
    const float* __restrict__ s0, const float* __restrict__ dinv,
    const int* __restrict__ jstar, const int* __restrict__ eflag,
    const float* __restrict__ score_b, float* __restrict__ score) {
  int row = blockIdx.x * 256 + threadIdx.x;
  if (row >= NN) return;
  float di = dinv[row];
  int j = jstar[row];
  float v = di * di * s0[row] + (eflag[row] ? di * dinv[j] * s0[j] : 0.0f) + score_b[0];
  score[row] = tanhf(v);
}

// ---------------- Kernel 4: exact top-k via LDS bitonic sort -----------------
__global__ __launch_bounds__(1024) void topk_kernel(
    const float* __restrict__ score, int* __restrict__ perm,
    float* __restrict__ tsc) {
  extern __shared__ unsigned long long keys[];
  const int tid = threadIdx.x;
  for (int i = tid; i < NN; i += 1024) {
    unsigned uo = __float_as_uint(score[i]);
    uo = (uo & 0x80000000u) ? ~uo : (uo | 0x80000000u);  // ascending float order
    unsigned du = ~uo;                                   // descending
    keys[i] = ((unsigned long long)du << 32) | (unsigned)i;
  }
  __syncthreads();
  for (int k = 2; k <= NN; k <<= 1) {
    for (int j = k >> 1; j > 0; j >>= 1) {
      for (int t = tid; t < NN; t += 1024) {
        int ixj = t ^ j;
        if (ixj > t) {
          bool up = ((t & k) == 0);
          unsigned long long a = keys[t], b = keys[ixj];
          if ((a > b) == up) { keys[t] = b; keys[ixj] = a; }
        }
      }
      __syncthreads();
    }
  }
  for (int i = tid; i < KKEEP; i += 1024) {
    int idx = (int)(keys[i] & 0xffffffffu);
    perm[i] = idx;
    tsc[i] = score[idx];
  }
}

// ---------------- Kernel 5: gather + LayerNorm + ReLU + head (f32) -----------
__global__ __launch_bounds__(256) void head_kernel(
    const float* __restrict__ g, const int* __restrict__ perm,
    const float* __restrict__ tsc, const float* __restrict__ ln_g,
    const float* __restrict__ ln_b, const float* __restrict__ head_W,
    const float* __restrict__ head_b, float* __restrict__ out) {
  __shared__ float hw[DH * DOUTC];
  __shared__ float rowbuf[8][DH];
  const int tid = threadIdx.x;
  for (int e = tid; e < DH * DOUTC; e += 256) hw[e] = head_W[e];
  __syncthreads();
  const int wave = tid >> 5, lane = tid & 31;
  const int base = (blockIdx.x * 8 + wave) * 8;
  for (int rr = 0; rr < 8; ++rr) {
    int k = base + rr;
    int p = perm[k];
    float ts = tsc[k];
    float x0 = g[p * DH + lane] * ts;
    float x1 = g[p * DH + lane + 32] * ts;
    float s = x0 + x1, q = x0 * x0 + x1 * x1;
#pragma unroll
    for (int d = 1; d < 32; d <<= 1) {
      s += __shfl_xor(s, d, 32);
      q += __shfl_xor(q, d, 32);
    }
    float mu = s * (1.0f / 64.0f);
    float var = q * (1.0f / 64.0f) - mu * mu;
    float rstd = rsqrtf(var + 1e-5f);
    float y0 = fmaxf((x0 - mu) * rstd * ln_g[lane] + ln_b[lane], 0.0f);
    float y1 = fmaxf((x1 - mu) * rstd * ln_g[lane + 32] + ln_b[lane + 32], 0.0f);
    rowbuf[wave][lane] = y0;
    rowbuf[wave][lane + 32] = y1;
    float acc = head_b[lane];
#pragma unroll
    for (int d = 0; d < DH; ++d) acc += rowbuf[wave][d] * hw[d * DOUTC + lane];
    out[k * DOUTC + lane] = acc;
  }
}

extern "C" void kernel_launch(void* const* d_in, const int* in_sizes, int n_in,
                              void* d_out, int out_size, void* d_ws, size_t ws_size,
                              hipStream_t stream) {
  (void)in_sizes; (void)n_in; (void)out_size; (void)ws_size;
  const float* x       = (const float*)d_in[0];
  const float* Wq      = (const float*)d_in[2];
  const float* bq      = (const float*)d_in[3];
  const float* Wk      = (const float*)d_in[4];
  const float* bk      = (const float*)d_in[5];
  const float* Wv      = (const float*)d_in[6];
  const float* bv      = (const float*)d_in[7];
  const float* gcn_W   = (const float*)d_in[8];
  const float* gcn_b   = (const float*)d_in[9];
  const float* score_W = (const float*)d_in[10];
  const float* score_b = (const float*)d_in[11];
  const float* ln_g    = (const float*)d_in[12];
  const float* ln_b    = (const float*)d_in[13];
  const float* head_W  = (const float*)d_in[14];
  const float* head_b  = (const float*)d_in[15];
  float* out = (float*)d_out;

  char* w = (char*)d_ws;
  auto take = [&](size_t bytes) { char* p = w; w += (bytes + 255) & ~(size_t)255; return p; };
  __bf16* Qb  = (__bf16*)take((size_t)NN * DH * 2);
  __bf16* Kb  = (__bf16*)take((size_t)NN * DH * 2);
  __bf16* VbT = (__bf16*)take((size_t)NN * DH * 2);
  float* h    = (float*)take((size_t)NN * DH * 4);
  float* u    = (float*)take((size_t)NN * DH * 4);
  float* g    = (float*)take((size_t)NN * DH * 4);
  float* dinv = (float*)take((size_t)NN * 4);
  int*   jst  = (int*)take((size_t)NN * 4);
  int*   efl  = (int*)take((size_t)NN * 4);
  float* s0   = (float*)take((size_t)NN * 4);
  float* sc   = (float*)take((size_t)NN * 4);
  int*   perm = (int*)take((size_t)KKEEP * 4);
  float* tsc  = (float*)take((size_t)KKEEP * 4);

  qkv_kernel<<<NN / 64, 256, 0, stream>>>(x, Wq, bq, Wk, bk, Wv, bv, Qb, Kb, VbT);
  attn_kernel<<<NN / 64, 128, 0, stream>>>(Qb, Kb, VbT, h, dinv, jst, efl);
  gcn_mm_kernel<<<NN / 128, 256, 0, stream>>>(h, gcn_W, u);
  gprop_kernel<<<NN / 8, 256, 0, stream>>>(u, dinv, jst, efl, gcn_b, score_W, g, s0);
  score_kernel<<<NN / 256, 256, 0, stream>>>(s0, dinv, jst, efl, score_b, sc);
  topk_kernel<<<1, 1024, NN * sizeof(unsigned long long), stream>>>(sc, perm, tsc);
  head_kernel<<<KKEEP / 64, 256, 0, stream>>>(g, perm, tsc, ln_g, ln_b, head_W, head_b, out);
}
